// L2DistanceTransformer_57346403336313
// MI455X (gfx1250) — matrix operand order repaired and checked by reading, not compiled, not run on hardware
//
#include <hip/hip_runtime.h>
#include <hip/hip_bf16.h>
#include <math.h>

// ---------------------------------------------------------------------------
// L2-distance transformer forward for MI455X (gfx1250, wave32, WMMA + TDM).
//
// Roofline: ~45 GFLOP total vs ~250 MB HBM traffic (131 MB of it the logits
// store) -> at 23.3 TB/s this is memory-bound (~11 us); compute is sub-us at
// any matrix-core rate. bf16 WMMA (f32 accumulate) for all GEMMs; exact fp32
// norm terms so only the cross term x.w passes through bf16. GEMM tiles are
// streamed global->LDS by the Tensor Data Mover (tensor_load_to_lds, D# per
// ISA ch.8), double-buffered on TENSORcnt, instead of per-thread staging.
// ---------------------------------------------------------------------------

#define N_TOK   1024
#define DIM     512
#define HEADS   8
#define DH      64
#define DI      512      // HEADS*DH
#define DFF     2048
#define VOCAB   32000
#define QKV_LD  1536     // 3*DI

typedef __attribute__((ext_vector_type(16))) __bf16 v16bf;
typedef __attribute__((ext_vector_type(8)))  __bf16 v8bf;
typedef __attribute__((ext_vector_type(8)))  float  v8f;
typedef __attribute__((ext_vector_type(4)))  unsigned u32x4;
typedef __attribute__((ext_vector_type(8)))  int      i32x8;
typedef __attribute__((ext_vector_type(4)))  int      i32x4;

__device__ __forceinline__ v8f wmma_bf16(v16bf a, v16bf b, v8f c) {
  // (neg_a, A, neg_b, B, c_mod, C, reuse_a, reuse_b)
  return __builtin_amdgcn_wmma_f32_16x16x32_bf16(false, a, false, b,
                                                 (short)0, c, false, false);
}

// Generic pointers to LDS carry the LDS byte offset in their low 32 bits
// (ISA 10.2).
__device__ __forceinline__ unsigned lds_off(const void* p) {
  return (unsigned)(unsigned long long)p;
}

// ---------------------------------------------------------------------------
// TDM: async-load one 64-row x 32-col fp32 tile (row stride ld elements) from
// global memory into LDS. D# built per CDNA5 ISA:
//  group0: count=1 | lds_addr | global_addr[56:0] | type=2
//  group1: data_size=4B, tensor_dim0=32, tensor_dim1=64, tile_dim0=32,
//          tile_dim1=64, tensor_dim0_stride=ld. Groups 2/3 zero (2D tile).
// Tracked on TENSORcnt; caller waits with s_wait_tensorcnt.
// ---------------------------------------------------------------------------
__device__ __forceinline__ void tdm_load_tile64x32_f32(unsigned lds_byte_off,
                                                       const float* gptr,
                                                       unsigned ld_elems) {
  unsigned long long ga = (unsigned long long)gptr;
  u32x4 g0;
  g0[0] = 1u;                                         // count=1, user mode
  g0[1] = lds_byte_off;                               // lds_addr
  g0[2] = (unsigned)(ga & 0xFFFFFFFFull);             // global_addr[31:0]
  g0[3] = (unsigned)((ga >> 32) & 0x1FFFFFFull)       // global_addr[56:32]
        | (2u << 30);                                 // type=2 ("image")
  i32x8 g1;
  g1[0] = (int)(2u << 16);          // workgroup_mask=0, data_size=2 (4 bytes)
  g1[1] = (int)(32u << 16);         // tensor_dim0 = 32 (lo16 in [31:16])
  g1[2] = (int)(64u << 16);         // tensor_dim0 hi=0 | tensor_dim1 = 64
  g1[3] = (int)(32u << 16);         // tensor_dim1 hi=0 | tile_dim0 = 32
  g1[4] = (int)64u;                 // tile_dim1 = 64, tile_dim2 = 0
  g1[5] = (int)ld_elems;            // tensor_dim0_stride[31:0]
  g1[6] = 0;                        // stride hi | tensor_dim1_stride lo
  g1[7] = 0;
  i32x4 z4 = {0, 0, 0, 0};
#if defined(__clang_major__) && (__clang_major__ >= 23)
  i32x8 z8 = {0, 0, 0, 0, 0, 0, 0, 0};
  __builtin_amdgcn_tensor_load_to_lds(g0, g1, z4, z4, z8, 0);
#else
  __builtin_amdgcn_tensor_load_to_lds(g0, g1, z4, z4, 0);
#endif
}

// ---------------------------------------------------------------------------
// x = token_emb[tokens] + pos_emb
// ---------------------------------------------------------------------------
__global__ __launch_bounds__(128)
void embed_kernel(const int* __restrict__ tokens, const float* __restrict__ temb,
                  const float* __restrict__ pemb, float* __restrict__ x) {
  int row = blockIdx.x;
  int tok = tokens[row];
  const float* te = temb + (long long)tok * DIM;
  const float* pe = pemb + row * DIM;
  float* xo = x + row * DIM;
  for (int d = threadIdx.x; d < DIM; d += 128) xo[d] = te[d] + pe[d];
}

// ---------------------------------------------------------------------------
// y = x/max(||x||,eps) * sqrt(DIM) * (gamma+1);  yss[row] = sum(y^2)
// ---------------------------------------------------------------------------
__global__ __launch_bounds__(128)
void rmsnorm_kernel(const float* __restrict__ x, const float* __restrict__ gamma,
                    float* __restrict__ y, float* __restrict__ yss) {
  __shared__ float sbuf[128];
  int row = blockIdx.x, t = threadIdx.x;
  const float* xr = x + row * DIM;
  float ss = 0.f;
  for (int d = t; d < DIM; d += 128) { float v = xr[d]; ss += v * v; }
  sbuf[t] = ss; __syncthreads();
  for (int s = 64; s > 0; s >>= 1) { if (t < s) sbuf[t] += sbuf[t + s]; __syncthreads(); }
  float nrm = sqrtf(sbuf[0]); __syncthreads();
  float scale = 22.627416997969522f / fmaxf(nrm, 1e-12f);   // sqrt(512)/||x||
  float ss2 = 0.f;
  float* yr = y + row * DIM;
  for (int d = t; d < DIM; d += 128) {
    float v = xr[d] * scale * (gamma[d] + 1.f);
    yr[d] = v; ss2 += v * v;
  }
  sbuf[t] = ss2; __syncthreads();
  for (int s = 64; s > 0; s >>= 1) { if (t < s) sbuf[t] += sbuf[t + s]; __syncthreads(); }
  if (t == 0) yss[row] = sbuf[0];
}

// ---------------------------------------------------------------------------
// out[row] = sum(w[row]^2) — exact fp32 weight-row norms for distance epilogues
// ---------------------------------------------------------------------------
__global__ __launch_bounds__(128)
void rowss_kernel(const float* __restrict__ w, float* __restrict__ out, int K) {
  __shared__ float sbuf[128];
  int row = blockIdx.x, t = threadIdx.x;
  const float* wr = w + (long long)row * K;
  float ss = 0.f;
  for (int d = t; d < K; d += 128) { float v = wr[d]; ss += v * v; }
  sbuf[t] = ss; __syncthreads();
  for (int s = 64; s > 0; s >>= 1) { if (t < s) sbuf[t] += sbuf[t + s]; __syncthreads(); }
  if (t == 0) out[row] = sbuf[0];
}

// ---------------------------------------------------------------------------
// WMMA GEMM:  dot = A[M,K] . W[N,K]^T  with fused distance epilogues.
// 64x64 tile per 256-thread workgroup (8 waves; each wave -> 16x32 of output).
// Tiles stream in via TDM (double-buffered on TENSORcnt); fp32->bf16 happens
// with native v_cvt while building WMMA fragments from LDS.
// ---------------------------------------------------------------------------
enum { EP_DOT_ADD = 0, EP_CDIST = 1, EP_NEG_SQDIST = 2, EP_GELU_FF = 3 };

template <int MODE>
__global__ __launch_bounds__(256)
void gemm_kernel(const float* __restrict__ A, const float* __restrict__ W,
                 float* __restrict__ C,
                 const float* __restrict__ xs, const float* __restrict__ ws,
                 const float* __restrict__ bias, const float* __restrict__ resid,
                 int M, int Nout, int K) {
  __shared__ float Asf[2][64][32];   // 8 KB per buffer
  __shared__ float Bsf[2][64][32];
  int t = threadIdx.x;
  int row0 = blockIdx.y * 64;
  int col0 = blockIdx.x * 64;
  int wave = t >> 5, lane = t & 31;
  int wm = (wave & 3) * 16;        // wave row offset in tile
  int wn = (wave >> 2) * 32;       // wave col offset in tile (2 subtiles)

  const float* Ag = A + (long long)row0 * K;
  const float* Wg = W + (long long)col0 * K;
  int nsteps = K >> 5;

  if (wave == 0) {  // prologue: tile 0 in flight (TDM ignores EXEC; 1 op/wave)
    tdm_load_tile64x32_f32(lds_off(&Asf[0][0][0]), Ag, (unsigned)K);
    tdm_load_tile64x32_f32(lds_off(&Bsf[0][0][0]), Wg, (unsigned)K);
  }

  v8f acc0 = {}, acc1 = {};
  for (int i = 0; i < nsteps; ++i) {
    int buf = i & 1;
    if (wave == 0) {
      if (i + 1 < nsteps) {  // stream next tile into the other buffer
        tdm_load_tile64x32_f32(lds_off(&Asf[buf ^ 1][0][0]), Ag + (i + 1) * 32,
                               (unsigned)K);
        tdm_load_tile64x32_f32(lds_off(&Bsf[buf ^ 1][0][0]), Wg + (i + 1) * 32,
                               (unsigned)K);
        __builtin_amdgcn_s_wait_tensorcnt(2);  // oldest pair (tile i) done
      } else {
        __builtin_amdgcn_s_wait_tensorcnt(0);
      }
    }
    __syncthreads();   // publish tile i to all waves

    // A operand: lane<16 -> M=lane, K 0..15; lane>=16 -> K 16..31
    int mrow = wm + (lane & 15);
    int c0 = (lane >> 4) * 16;
    const float* ar = &Asf[buf][mrow][0];
    v16bf afr;
#pragma unroll
    for (int j = 0; j < 8; ++j) afr[j] = (__bf16)ar[c0 + j];
#pragma unroll
    for (int j = 0; j < 8; ++j) afr[j + 8] = (__bf16)ar[c0 + 8 + j];
    // B operand: lane holds one output column, contiguous 16 K-values
    int kk = (lane >> 4) * 16;
    int nb = wn + (lane & 15);
    const float* br0 = &Bsf[buf][nb][0];
    const float* br1 = &Bsf[buf][nb + 16][0];
    v16bf bfr0, bfr1;
#pragma unroll
    for (int j = 0; j < 16; ++j) bfr0[j] = (__bf16)br0[kk + j];
#pragma unroll
    for (int j = 0; j < 16; ++j) bfr1[j] = (__bf16)br1[kk + j];
    acc0 = wmma_bf16(afr, bfr0, acc0);
    acc1 = wmma_bf16(afr, bfr1, acc1);
    __syncthreads();   // tile i consumed; its buffer may be overwritten next
  }

  // Epilogue. D layout: lane = N (+16*(M/8) half), vgpr r -> M = r + 8*(lane/16).
  int nloc = lane & 15;
#pragma unroll
  for (int sub = 0; sub < 2; ++sub) {
    v8f acc = sub ? acc1 : acc0;
    int ncol = col0 + wn + sub * 16 + nloc;
#pragma unroll
    for (int r = 0; r < 8; ++r) {
      int mrow2 = row0 + wm + r + 8 * (lane >> 4);
      float dot = acc[r];
      float outv;
      if (MODE == EP_DOT_ADD) {
        outv = dot + resid[(long long)mrow2 * Nout + ncol];
      } else if (MODE == EP_CDIST) {
        float d2 = xs[mrow2] - 2.f * dot + ws[ncol];
        outv = sqrtf(fmaxf(d2, 0.f));
      } else if (MODE == EP_NEG_SQDIST) {
        outv = -(xs[mrow2] - 2.f * dot + ws[ncol]);
      } else { // EP_GELU_FF: gelu(-sqdist + bias), exact erf gelu
        float v = -(xs[mrow2] - 2.f * dot + ws[ncol]) + bias[ncol];
        outv = 0.5f * v * (1.f + erff(v * 0.70710678118654752f));
      }
      C[(long long)mrow2 * Nout + ncol] = outv;
    }
  }
}

// ---------------------------------------------------------------------------
// Fused distance attention. One 128-thread (4-wave) block per (16 q-rows, head).
// sim = (2 q.k - ||q||^2 - ||k||^2) / 8 ; softmax over all 1024 keys ; O = P.V
// Scores kept as bf16 in LDS (exactly the WMMA P operand); 1/rowsum folded at
// the end.
// ---------------------------------------------------------------------------
__global__ __launch_bounds__(128)
void attn_kernel(const float* __restrict__ qkv, float* __restrict__ attno) {
  __shared__ __bf16 S16[16][1024];           // 32 KB scores/P
  __shared__ float Obuf[4][16][64];          // 16 KB per-wave partial O
  __shared__ float qs_s[16];
  __shared__ float rsum[16];

  int i0 = blockIdx.x * 16;
  int h = blockIdx.y;
  int t = threadIdx.x, wave = t >> 5, lane = t & 31;
  int qoff = h * DH, koff = DI + h * DH, voff = 2 * DI + h * DH;

  { // ||q||^2 per row (8 threads/row, shfl tree within 8-lane groups)
    int r = t >> 3, k8 = t & 7;
    const float* p = qkv + (i0 + r) * QKV_LD + qoff;
    float s = 0.f;
    for (int d = k8; d < DH; d += 8) { float v = p[d]; s += v * v; }
    s += __shfl_xor(s, 4, 32); s += __shfl_xor(s, 2, 32); s += __shfl_xor(s, 1, 32);
    if (k8 == 0) qs_s[r] = s;
  }
  __syncthreads();

  // ---- scores: each wave owns disjoint 16-wide j tiles
  for (int it = 0; it < 16; ++it) {
    int j0 = (it * 4 + wave) * 16;
    float ksn;
    { // ||k_{j0 + lane%16}||^2 (dup in upper lanes; matches D layout need)
      const float* p = qkv + (j0 + (lane & 15)) * QKV_LD + koff;
      float s = 0.f;
      for (int d = 0; d < DH; d += 4) {
        float4 v = *(const float4*)(p + d);
        s += v.x * v.x + v.y * v.y + v.z * v.z + v.w * v.w;
      }
      ksn = s;
    }
    v8f acc = {};
#pragma unroll
    for (int kb = 0; kb < 2; ++kb) {       // K = 64 = 2 x 32
      int kbase = kb * 32;
      v16bf afr;
      { // Q operand (fp32 -> bf16 native cvt in regs)
        const float* p = qkv + (i0 + (lane & 15)) * QKV_LD + qoff + kbase + ((lane >> 4) * 16);
#pragma unroll
        for (int j = 0; j < 8; ++j) afr[j] = (__bf16)p[j];
#pragma unroll
        for (int j = 0; j < 8; ++j) afr[j + 8] = (__bf16)p[8 + j];
      }
      v16bf bfr;
      { // K^T operand: lane = key j0+n, 16 contiguous dims
        const float* p = qkv + (j0 + (lane & 15)) * QKV_LD + koff + kbase + ((lane >> 4) * 16);
#pragma unroll
        for (int j = 0; j < 16; ++j) bfr[j] = (__bf16)p[j];
      }
      acc = wmma_bf16(afr, bfr, acc);
    }
    { // sim epilogue -> LDS bf16
      int n = lane & 15;
#pragma unroll
      for (int r = 0; r < 8; ++r) {
        int m = r + 8 * (lane >> 4);
        float sim = (2.f * acc[r] - qs_s[m] - ksn) * 0.125f;  // * DH^-0.5
        S16[m][j0 + n] = (__bf16)sim;
      }
    }
  }
  __syncthreads();

  { // softmax (store unnormalized exp as bf16 = P operand; keep row sums)
    int r = t >> 3, k8 = t & 7;
    float mx = -3.0e38f;
    for (int j = k8; j < N_TOK; j += 8) mx = fmaxf(mx, (float)S16[r][j]);
    mx = fmaxf(mx, __shfl_xor(mx, 4, 32));
    mx = fmaxf(mx, __shfl_xor(mx, 2, 32));
    mx = fmaxf(mx, __shfl_xor(mx, 1, 32));
    float sm = 0.f;
    for (int j = k8; j < N_TOK; j += 8) {
      float e = __expf((float)S16[r][j] - mx);
      S16[r][j] = (__bf16)e;
      sm += e;
    }
    sm += __shfl_xor(sm, 4, 32); sm += __shfl_xor(sm, 2, 32); sm += __shfl_xor(sm, 1, 32);
    if (k8 == 0) rsum[r] = sm;
  }
  __syncthreads();

  // ---- O = P.V: each wave reduces its 256-key slice, K steps of 32
  v8f o[4] = {{}, {}, {}, {}};
  int jbase = wave * 256;
  for (int kb = 0; kb < 8; ++kb) {
    int kj = jbase + kb * 32;
    int c0 = (lane >> 4) * 16;
    const v8bf* plo = (const v8bf*)&S16[lane & 15][kj + c0];
    const v8bf* phi = (const v8bf*)&S16[lane & 15][kj + c0 + 8];
    v8bf lo = *plo, hi = *phi;
    v16bf afr;
#pragma unroll
    for (int j = 0; j < 8; ++j) { afr[j] = lo[j]; afr[j + 8] = hi[j]; }
#pragma unroll
    for (int sub = 0; sub < 4; ++sub) {    // 64 output dims = 4 x 16
      int d = sub * 16 + (lane & 15);
      int jr = kj + (lane >> 4) * 16;
      v16bf bfr;
#pragma unroll
      for (int j = 0; j < 16; ++j) bfr[j] = (__bf16)qkv[(jr + j) * QKV_LD + voff + d];
      o[sub] = wmma_bf16(afr, bfr, o[sub]);
    }
  }
  { // spill per-wave partials
    int n = lane & 15;
#pragma unroll
    for (int sub = 0; sub < 4; ++sub)
#pragma unroll
      for (int r = 0; r < 8; ++r)
        Obuf[wave][r + 8 * (lane >> 4)][sub * 16 + n] = o[sub][r];
  }
  __syncthreads();
  for (int e = t; e < 16 * 64; e += 128) {  // cross-wave reduce + 1/rowsum
    int m = e >> 6, d = e & 63;
    float s = Obuf[0][m][d] + Obuf[1][m][d] + Obuf[2][m][d] + Obuf[3][m][d];
    attno[(i0 + m) * DI + h * DH + d] = s / rsum[m];
  }
}

// ---------------------------------------------------------------------------
extern "C" void kernel_launch(void* const* d_in, const int* in_sizes, int n_in,
                              void* d_out, int out_size, void* d_ws, size_t ws_size,
                              hipStream_t stream) {
  (void)in_sizes; (void)n_in; (void)out_size;
  const int*   tokens      = (const int*)d_in[0];
  const float* token_emb   = (const float*)d_in[1];
  const float* pos_emb     = (const float*)d_in[2];
  const float* qkv_w       = (const float*)d_in[3];
  const float* attn_out_w  = (const float*)d_in[4];
  const float* ff_in_w     = (const float*)d_in[5];
  const float* ff_in_b     = (const float*)d_in[6];
  const float* ff_out_w    = (const float*)d_in[7];
  const float* attn_gamma  = (const float*)d_in[8];
  const float* ff_gamma    = (const float*)d_in[9];
  const float* final_gamma = (const float*)d_in[10];
  float* out = (float*)d_out;
  float* ws  = (float*)d_ws;

  // workspace carve-up (floats)
  float* x     = ws;                       // N*DIM
  float* y     = x + N_TOK * DIM;          // N*DIM
  float* yss   = y + N_TOK * DIM;          // N
  float* qkv   = yss + N_TOK;              // N*1536
  float* attno = qkv + N_TOK * QKV_LD;     // N*DI
  float* hbuf  = attno + N_TOK * DI;       // N*DFF
  float* wss   = hbuf + N_TOK * DFF;       // VOCAB (max weight-row count)
  size_t need = (size_t)(wss + VOCAB - ws) * sizeof(float);   // ~21.2 MB
  if (ws_size < need) return;              // refuse to scribble out of bounds

  dim3 tpb256(256), tpb128(128);

  embed_kernel<<<N_TOK, tpb128, 0, stream>>>(tokens, token_emb, pos_emb, x);

  for (int l = 0; l < 2; ++l) {
    const float* qw  = qkv_w + (long long)l * QKV_LD * DIM;
    const float* ow  = attn_out_w + (long long)l * DIM * DI;
    const float* fiw = ff_in_w + (long long)l * DFF * DIM;
    const float* fib = ff_in_b + (long long)l * DFF;
    const float* fow = ff_out_w + (long long)l * DIM * DFF;

    // x = attention(rmsnorm(x)) + x
    rmsnorm_kernel<<<N_TOK, tpb128, 0, stream>>>(x, attn_gamma + l * DIM, y, yss);
    rowss_kernel<<<QKV_LD, tpb128, 0, stream>>>(qw, wss, DIM);
    gemm_kernel<EP_CDIST><<<dim3(QKV_LD / 64, N_TOK / 64), tpb256, 0, stream>>>(
        y, qw, qkv, yss, wss, nullptr, nullptr, N_TOK, QKV_LD, DIM);
    attn_kernel<<<dim3(N_TOK / 16, HEADS), tpb128, 0, stream>>>(qkv, attno);
    gemm_kernel<EP_DOT_ADD><<<dim3(DIM / 64, N_TOK / 64), tpb256, 0, stream>>>(
        attno, ow, x, nullptr, nullptr, nullptr, x, N_TOK, DIM, DI);

    // x = ff(rmsnorm(x)) + x
    rmsnorm_kernel<<<N_TOK, tpb128, 0, stream>>>(x, ff_gamma + l * DIM, y, yss);
    rowss_kernel<<<DFF, tpb128, 0, stream>>>(fiw, wss, DIM);
    gemm_kernel<EP_GELU_FF><<<dim3(DFF / 64, N_TOK / 64), tpb256, 0, stream>>>(
        y, fiw, hbuf, yss, wss, fib, nullptr, N_TOK, DFF, DIM);
    gemm_kernel<EP_DOT_ADD><<<dim3(DIM / 64, N_TOK / 64), tpb256, 0, stream>>>(
        hbuf, fow, x, nullptr, nullptr, nullptr, x, N_TOK, DIM, DFF);
  }

  // logits = -sq_dist(rmsnorm(x), token_emb)
  rmsnorm_kernel<<<N_TOK, tpb128, 0, stream>>>(x, final_gamma, y, yss);
  rowss_kernel<<<VOCAB, tpb128, 0, stream>>>(token_emb, wss, DIM);
  gemm_kernel<EP_NEG_SQDIST><<<dim3(VOCAB / 64, N_TOK / 64), tpb256, 0, stream>>>(
      y, token_emb, out, yss, wss, nullptr, nullptr, N_TOK, VOCAB, DIM);
}
